// PointCloudEncoder_14001593385268
// MI455X (gfx1250) — matrix operand order repaired and checked by reading, not compiled, not run on hardware
//
#include <hip/hip_runtime.h>
#include <hip/hip_bf16.h>

typedef _Float16 h16;
typedef _Float16 v16h __attribute__((ext_vector_type(16)));
typedef _Float16 v8h  __attribute__((ext_vector_type(8)));
typedef float    v8f  __attribute__((ext_vector_type(8)));

// ---------------------------------------------------------------------------
// FPS: single workgroup, 256 threads. dist[] lives in registers (32 items per
// thread), argmax via LDS tree reduce. Emits the reference's exact sequence:
// out[i] = far (pre-update), then dist=min(dist,d(far)), far=argmax(dist).
// ---------------------------------------------------------------------------
__global__ void k_fps(const float* __restrict__ xyz, int N, int npoint, int* __restrict__ out)
{
    const int T = 256;
    const int MAXIT = 32; // N <= 8192
    __shared__ float rd[T];
    __shared__ int   ri[T];
    int t = threadIdx.x;
    float dist[MAXIT];
#pragma unroll
    for (int it = 0; it < MAXIT; ++it) dist[it] = 1e10f;
    int far = 0;
    for (int i = 0; i < npoint; ++i) {
        if (t == 0) out[i] = far;
        float px = xyz[far*3+0], py = xyz[far*3+1], pz = xyz[far*3+2];
        float best = -1.0f; int bidx = 0x7fffffff;
#pragma unroll
        for (int it = 0; it < MAXIT; ++it) {
            int idx = t + it * T;
            if (idx < N) {
                float dx = xyz[idx*3+0]-px, dy = xyz[idx*3+1]-py, dz = xyz[idx*3+2]-pz;
                float d = dx*dx + dy*dy + dz*dz;
                dist[it] = fminf(dist[it], d);
                if (dist[it] > best || (dist[it] == best && idx < bidx)) { best = dist[it]; bidx = idx; }
            }
        }
        rd[t] = best; ri[t] = bidx;
        __syncthreads();
        for (int s = T/2; s > 0; s >>= 1) {
            if (t < s) {
                float od = rd[t+s]; int oi = ri[t+s];
                if (od > rd[t] || (od == rd[t] && oi < ri[t])) { rd[t] = od; ri[t] = oi; }
            }
            __syncthreads();
        }
        far = ri[0];
        __syncthreads();
    }
}

__global__ void k_cls0(int* __restrict__ cls, int n)
{
    int i = blockIdx.x * 256 + threadIdx.x;
    if (i < n) cls[i] = i >> 11; // 2048 points per class
}

__global__ void k_gather_level(const float* __restrict__ xi, const int* __restrict__ ci,
                               const int* __restrict__ fi, float* __restrict__ xo,
                               int* __restrict__ co, int np)
{
    int i = blockIdx.x * 128 + threadIdx.x;
    if (i >= np) return;
    int s = fi[i];
    xo[i*3+0] = xi[s*3+0]; xo[i*3+1] = xi[s*3+1]; xo[i*3+2] = xi[s*3+2];
    co[i] = ci[s];
}

// ---------------------------------------------------------------------------
// Ball query: one wave per center, first 32 same-class hits with d^2 <= r^2
// in index order (cross-class reference distance is 1.0 > r^2), padded with
// the first hit.
// ---------------------------------------------------------------------------
__global__ void k_query_ball(const float* __restrict__ xyz, const int* __restrict__ cls,
                             const int* __restrict__ fi, int N, int np, float r2,
                             int* __restrict__ gi)
{
    int w = blockIdx.x * blockDim.y + threadIdx.y;
    if (w >= np) return;
    int lane = threadIdx.x;
    int c = fi[w];
    float cx = xyz[c*3+0], cy = xyz[c*3+1], cz = xyz[c*3+2];
    int ccls = cls[c];
    int count = 0, first = -1;
    for (int j0 = 0; j0 < N && count < 32; j0 += 32) {
        int j = j0 + lane;
        bool ok = false;
        if (j < N && cls[j] == ccls) {
            float dx = xyz[j*3+0]-cx, dy = xyz[j*3+1]-cy, dz = xyz[j*3+2]-cz;
            ok = (dx*dx + dy*dy + dz*dz) <= r2;
        }
        unsigned mask = (unsigned)__ballot(ok);
        if (first < 0 && mask) first = j0 + __ffs(mask) - 1;
        if (ok) {
            int pos = count + __popc(mask & ((1u << lane) - 1u));
            if (pos < 32) gi[w*32 + pos] = j;
        }
        count += __popc(mask);
    }
    if (count < 32) {
        for (int p = count + lane; p < 32; p += 32) gi[w*32 + p] = first;
    }
}

// ---------------------------------------------------------------------------
// Grouped SA input rows: A[(i*32+s), :] = [xyz[g]-xyz[center], feat[g]],
// zero-padded to Kp (multiple of 32), f16.
// ---------------------------------------------------------------------------
__global__ void k_group(const float* __restrict__ xyz, const int* __restrict__ fi,
                        const int* __restrict__ gi,
                        const float* __restrict__ verts, const float* __restrict__ normals,
                        const h16* __restrict__ pts, int Cfeat,
                        h16* __restrict__ A, int Kp, int np)
{
    int w = blockIdx.x * blockDim.y + threadIdx.y;
    if (w >= np * 32) return;
    int i = w >> 5;
    int g = gi[w];
    int c0 = fi[i];
    int lane = threadIdx.x;
    float cx = xyz[c0*3+0], cy = xyz[c0*3+1], cz = xyz[c0*3+2];
    for (int c = lane; c < Kp; c += 32) {
        float v = 0.0f;
        if (c == 0)      v = xyz[g*3+0] - cx;
        else if (c == 1) v = xyz[g*3+1] - cy;
        else if (c == 2) v = xyz[g*3+2] - cz;
        else if (pts) { int cc = c - 3; if (cc < Cfeat) v = (float)pts[(size_t)g*Cfeat + cc]; }
        else {
            int cc = c - 3;
            if (cc < 3)      v = verts[g*3+cc];
            else if (cc < 6) v = normals[g*3 + (cc-3)];
        }
        A[(size_t)w*Kp + c] = (h16)v;
    }
}

// ---------------------------------------------------------------------------
// WMMA GEMM with async-LDS weight staging.
//   C[M,Nn] = act(A[M,Kp] @ B[Kp,Nn] + bias)
// A: row-major f16, Kp%32==0 zero padded. Bt: weights pre-transposed (Nn,Kp)
// f16, so a 64-column strip is one contiguous region -> staged into LDS once
// per block with global_load_async_to_lds_b128 (ASYNCcnt) and served to the
// k-loop via ds_load. Each wave computes a 2x2 grid of 16x16 WMMA tiles
// (4 v_wmma per 4 A-fragment loads); 4 waves tile a 64x64 output block.
// ---------------------------------------------------------------------------
__global__ void k_gemm(const h16* __restrict__ A, const h16* __restrict__ Bt,
                       const float* __restrict__ bias, void* __restrict__ out,
                       int M, int Nn, int Kp, int relu, int outf32)
{
    extern __shared__ h16 sB[]; // 64 rows (output cols) x Kp halves
    const int tid = threadIdx.y * 32 + threadIdx.x;
    const int by = blockIdx.y;

    // ---- async-stage the contiguous 64xKp weight strip into LDS ----
    {
        unsigned long long gbase = (unsigned long long)(Bt + (size_t)by * 64 * Kp);
        unsigned lbase = (unsigned)(unsigned long long)&sB[0];
        int nchunk = (64 * Kp) / 8; // 16B chunks; Kp%32==0 -> nchunk%256==0
        for (int c = tid; c < nchunk; c += 128) {
            unsigned long long ga = gbase + (unsigned long long)c * 16ull;
            unsigned la = lbase + (unsigned)c * 16u;
            asm volatile("global_load_async_to_lds_b128 %0, %1, off"
                         :: "v"(la), "v"(ga) : "memory");
        }
        asm volatile("s_wait_asynccnt 0" ::: "memory");
    }
    __syncthreads();

    const int lane = threadIdx.x;
    const int half = lane >> 4;
    const int l16  = lane & 15;
    const int wm   = threadIdx.y >> 1;
    const int wn   = threadIdx.y & 1;
    const long row0 = (long)blockIdx.x * 64 + wm * 32;
    const int  nl0  = wn * 32;

    const h16* arow0 = A + (row0 + l16)      * Kp + 16 * half;
    const h16* arow1 = A + (row0 + 16 + l16) * Kp + 16 * half;
    const int  boff0 = (nl0 + l16)      * Kp + 16 * half;
    const int  boff1 = (nl0 + 16 + l16) * Kp + 16 * half;

    v8f acc00 = {}, acc01 = {}, acc10 = {}, acc11 = {};
    for (int k = 0; k < Kp; k += 32) {
        v8h a0lo = *(const v8h*)(arow0 + k);
        v8h a0hi = *(const v8h*)(arow0 + k + 8);
        v8h a1lo = *(const v8h*)(arow1 + k);
        v8h a1hi = *(const v8h*)(arow1 + k + 8);
        v16h a0, a1;
#pragma unroll
        for (int q = 0; q < 8; ++q) {
            a0[q] = a0lo[q]; a0[q+8] = a0hi[q];
            a1[q] = a1lo[q]; a1[q+8] = a1hi[q];
        }
        v16h b0 = *(const v16h*)&sB[boff0 + k];
        v16h b1 = *(const v16h*)&sB[boff1 + k];
        acc00 = __builtin_amdgcn_wmma_f32_16x16x32_f16(false, a0, false, b0, (short)0, acc00, false, false);
        acc01 = __builtin_amdgcn_wmma_f32_16x16x32_f16(false, a0, false, b1, (short)0, acc01, false, false);
        acc10 = __builtin_amdgcn_wmma_f32_16x16x32_f16(false, a1, false, b0, (short)0, acc10, false, false);
        acc11 = __builtin_amdgcn_wmma_f32_16x16x32_f16(false, a1, false, b1, (short)0, acc11, false, false);
    }

    long col0 = (long)by * 64 + nl0 + l16;
    long col1 = col0 + 16;
    float bb0 = bias ? bias[col0] : 0.0f;
    float bb1 = bias ? bias[col1] : 0.0f;
#pragma unroll
    for (int r = 0; r < 8; ++r) {
        long r0 = row0 + r + 8*half;
        long r1 = r0 + 16;
        float v00 = acc00[r] + bb0, v01 = acc01[r] + bb1;
        float v10 = acc10[r] + bb0, v11 = acc11[r] + bb1;
        if (relu) {
            v00 = v00 > 0.0f ? v00 : 0.0f; v01 = v01 > 0.0f ? v01 : 0.0f;
            v10 = v10 > 0.0f ? v10 : 0.0f; v11 = v11 > 0.0f ? v11 : 0.0f;
        }
        if (outf32) {
            float* o = (float*)out;
            o[r0*Nn + col0] = v00; o[r0*Nn + col1] = v01;
            o[r1*Nn + col0] = v10; o[r1*Nn + col1] = v11;
        } else {
            h16* o = (h16*)out;
            o[r0*Nn + col0] = (h16)v00; o[r0*Nn + col1] = (h16)v01;
            o[r1*Nn + col0] = (h16)v10; o[r1*Nn + col1] = (h16)v11;
        }
    }
}

__global__ void k_maxpool(const h16* __restrict__ in, h16* __restrict__ out, int np, int C)
{
    int idx = blockIdx.x * 256 + threadIdx.x;
    if (idx >= np * C) return;
    int i = idx / C, c = idx - i * C;
    const h16* base = in + (size_t)i * 32 * C + c;
    float m = -3.4e38f;
#pragma unroll 4
    for (int s = 0; s < 32; ++s) m = fmaxf(m, (float)base[(size_t)s * C]);
    out[idx] = (h16)m;
}

// ---------------------------------------------------------------------------
// FP stage input: one wave per point. Top-3 nearest (cross-class d = 1.0,
// ties -> lower index like jax top_k), inverse-distance weights, writes
// A row = [points1 | interp], zero-padded to Kp, f16.
// ---------------------------------------------------------------------------
__global__ void k_fp_build(const float* __restrict__ xyz1, const int* __restrict__ cls1, int N1,
                           const float* __restrict__ xyz2, const int* __restrict__ cls2, int N2,
                           const h16* __restrict__ p1, int C1,
                           const h16* __restrict__ p2, int C2,
                           h16* __restrict__ A, int Kp)
{
    int i = blockIdx.x * blockDim.y + threadIdx.y;
    if (i >= N1) return;
    int lane = threadIdx.x;
    float cx = xyz1[i*3+0], cy = xyz1[i*3+1], cz = xyz1[i*3+2];
    int ci = cls1[i];
    int sel[3] = {-1, -1, -1};
    float sd[3];
    for (int r = 0; r < 3; ++r) {
        float bd = 3.4e38f; int bi = 0x7fffffff;
        for (int j = lane; j < N2; j += 32) {
            if (j == sel[0] || j == sel[1] || j == sel[2]) continue;
            float d;
            if (cls2[j] == ci) {
                float dx = xyz2[j*3+0]-cx, dy = xyz2[j*3+1]-cy, dz = xyz2[j*3+2]-cz;
                d = dx*dx + dy*dy + dz*dz;
            } else d = 1.0f;
            if (d < bd || (d == bd && j < bi)) { bd = d; bi = j; }
        }
        for (int off = 16; off > 0; off >>= 1) {
            float od = __shfl_xor(bd, off);
            int   oi = __shfl_xor(bi, off);
            if (od < bd || (od == bd && oi < bi)) { bd = od; bi = oi; }
        }
        sel[r] = bi; sd[r] = bd;
    }
    float w0 = 1.0f / (sd[0] + 1e-8f);
    float w1 = 1.0f / (sd[1] + 1e-8f);
    float w2 = 1.0f / (sd[2] + 1e-8f);
    float inv = 1.0f / (w0 + w1 + w2);
    w0 *= inv; w1 *= inv; w2 *= inv;
    for (int c = lane; c < Kp; c += 32) {
        float v = 0.0f;
        if (c < C1) v = (float)p1[(size_t)i*C1 + c];
        else if (c < C1 + C2) {
            int cc = c - C1;
            v = w0 * (float)p2[(size_t)sel[0]*C2 + cc]
              + w1 * (float)p2[(size_t)sel[1]*C2 + cc]
              + w2 * (float)p2[(size_t)sel[2]*C2 + cc];
        }
        A[(size_t)i*Kp + c] = (h16)v;
    }
}

// Weights: f32 (K,N) -> f16 transposed (N, Kp), zero-padded along K.
__global__ void k_prep_w(const float* __restrict__ W, h16* __restrict__ Wt, int K, int N, int Kp)
{
    int idx = blockIdx.x * 256 + threadIdx.x;
    if (idx >= N * Kp) return;
    int n = idx / Kp, k = idx - n * Kp;
    Wt[idx] = (k < K) ? (h16)W[(size_t)k * N + n] : (h16)0.0f;
}

__global__ void k_colstats(const float* __restrict__ X, int M, int C, float* __restrict__ st)
{
    __shared__ float s1[256], s2[256];
    int c = blockIdx.x, t = threadIdx.x;
    float a = 0.0f, b = 0.0f;
    for (int r = t; r < M; r += 256) { float v = X[(size_t)r*C + c]; a += v; b += v*v; }
    s1[t] = a; s2[t] = b;
    __syncthreads();
    for (int s = 128; s > 0; s >>= 1) {
        if (t < s) { s1[t] += s1[t+s]; s2[t] += s2[t+s]; }
        __syncthreads();
    }
    if (t == 0) {
        float mu = s1[0] / M;
        st[2*c]   = mu;
        st[2*c+1] = s2[0] / M - mu * mu;
    }
}

__global__ void k_norm(const float* __restrict__ X, const float* __restrict__ st,
                       float* __restrict__ out, int M, int C)
{
    int idx = blockIdx.x * 256 + threadIdx.x;
    if (idx >= M * C) return;
    int c = idx % C;
    float v = (X[idx] - st[2*c]) * rsqrtf(st[2*c+1] + 1e-5f);
    out[idx] = v > 0.0f ? v : 0.0f;
}

// ---------------------------------------------------------------------------
extern "C" void kernel_launch(void* const* d_in, const int* in_sizes, int n_in,
                              void* d_out, int out_size, void* d_ws, size_t ws_size,
                              hipStream_t stream)
{
    (void)in_sizes; (void)n_in; (void)out_size; (void)ws_size;
    const float* verts   = (const float*)d_in[0]; // (8192,3)
    const float* normals = (const float*)d_in[1]; // (8192,3)

    auto rnd32 = [](int x) { return (x + 31) & ~31; };
    char* base = (char*)d_ws; size_t off = 0;
    auto alloc = [&](size_t bytes) -> void* {
        void* p = base + off; off = (off + bytes + 255) & ~(size_t)255; return p;
    };

    // ---- parameter walk (setup_inputs insertion order: sa, fp, conv1) ----
    struct Lyr { const float* W; const float* b; int K, N, Kp; h16* Wt; };
    static const int sa_d[4][4] = {{9,64,64,128},{131,128,128,256},{259,256,256,512},{515,512,512,1024}};
    static const int fp_d[4][4] = {{1536,768,512,0},{768,512,256,0},{384,256,128,0},{128,128,128,64}};
    static const int fp_n[4] = {2,2,2,3};
    Lyr sa[4][3], fp[4][3], cv;
    int p = 2;
    for (int l = 0; l < 4; ++l)
        for (int j = 0; j < 3; ++j) {
            sa[l][j].K = sa_d[l][j]; sa[l][j].N = sa_d[l][j+1]; sa[l][j].Kp = rnd32(sa_d[l][j]);
            sa[l][j].W = (const float*)d_in[p++]; sa[l][j].b = (const float*)d_in[p++];
        }
    for (int l = 0; l < 4; ++l)
        for (int j = 0; j < fp_n[l]; ++j) {
            fp[l][j].K = fp_d[l][j]; fp[l][j].N = fp_d[l][j+1]; fp[l][j].Kp = rnd32(fp_d[l][j]);
            fp[l][j].W = (const float*)d_in[p++]; fp[l][j].b = (const float*)d_in[p++];
        }
    cv.K = 64; cv.N = 64; cv.Kp = 64;
    cv.W = (const float*)d_in[p++]; cv.b = (const float*)d_in[p++];

    auto prep = [&](Lyr& L) {
        L.Wt = (h16*)alloc((size_t)L.N * L.Kp * sizeof(h16));
        int tot = L.N * L.Kp;
        k_prep_w<<<(tot + 255) / 256, 256, 0, stream>>>(L.W, L.Wt, L.K, L.N, L.Kp);
    };
    for (int l = 0; l < 4; ++l) for (int j = 0; j < 3; ++j)       prep(sa[l][j]);
    for (int l = 0; l < 4; ++l) for (int j = 0; j < fp_n[l]; ++j) prep(fp[l][j]);
    prep(cv);

    // ---- workspace ----
    const int NL[5] = {8192, 2048, 512, 128, 32};
    const float R2[4] = {0.06f*0.06f, 0.1f*0.1f, 0.14f*0.14f, 0.18f*0.18f};
    const size_t BUF = (size_t)65536 * 128 * sizeof(h16);
    h16* bufA = (h16*)alloc(BUF);
    h16* bufB = (h16*)alloc(BUF);
    const float* XYZ[5]; float* xyzL[5]; int* clsL[5]; int* fL[4]; int* gL[4];
    XYZ[0] = verts;
    clsL[0] = (int*)alloc(NL[0] * sizeof(int));
    for (int k = 1; k < 5; ++k) {
        xyzL[k] = (float*)alloc((size_t)NL[k] * 3 * sizeof(float));
        clsL[k] = (int*)alloc(NL[k] * sizeof(int));
        XYZ[k] = xyzL[k];
    }
    for (int k = 0; k < 4; ++k) {
        fL[k] = (int*)alloc(NL[k+1] * sizeof(int));
        gL[k] = (int*)alloc((size_t)NL[k+1] * 32 * sizeof(int));
    }
    h16* ptsS[5] = {nullptr};
    for (int k = 0; k < 4; ++k) ptsS[k+1] = (h16*)alloc((size_t)NL[k+1] * sa_d[k][3] * sizeof(h16));
    h16* ptsF3 = (h16*)alloc((size_t)128 * 512 * sizeof(h16));
    h16* ptsF2 = (h16*)alloc((size_t)512 * 256 * sizeof(h16));
    h16* ptsF1 = (h16*)alloc((size_t)2048 * 128 * sizeof(h16));
    h16* l0feat = (h16*)alloc((size_t)8192 * 64 * sizeof(h16));
    float* Xbuf = (float*)alloc((size_t)8192 * 64 * sizeof(float));
    float* stats = (float*)alloc(64 * 2 * sizeof(float));

    auto gemm = [&](const h16* A, const Lyr& L, void* out, int M, int Kp, int relu, int outf32) {
        dim3 blk(32, 4), grd(M / 64, L.N / 64);
        size_t smem = (size_t)64 * Kp * sizeof(h16);
        k_gemm<<<grd, blk, smem, stream>>>(A, L.Wt, L.b, out, M, L.N, Kp, relu, outf32);
    };

    k_cls0<<<(NL[0] + 255) / 256, 256, 0, stream>>>(clsL[0], NL[0]);

    // ---- SA stages ----
    for (int k = 0; k < 4; ++k) {
        int N = NL[k], np = NL[k+1];
        k_fps<<<1, 256, 0, stream>>>(XYZ[k], N, np, fL[k]);
        k_gather_level<<<(np + 127) / 128, 128, 0, stream>>>(XYZ[k], clsL[k], fL[k],
                                                             xyzL[k+1], clsL[k+1], np);
        k_query_ball<<<(np + 3) / 4, dim3(32, 4), 0, stream>>>(XYZ[k], clsL[k], fL[k],
                                                               N, np, R2[k], gL[k]);
        int Kp = sa[k][0].Kp;
        int Cfeat = (k == 0) ? 0 : sa_d[k-1][3];
        int rows = np * 32;
        k_group<<<rows / 8, dim3(32, 8), 0, stream>>>(XYZ[k], fL[k], gL[k], verts, normals,
                                                      (k == 0) ? nullptr : ptsS[k], Cfeat,
                                                      bufA, Kp, np);
        const h16* in = bufA;
        for (int j = 0; j < 3; ++j) {
            h16* ob = (in == bufA) ? bufB : bufA;
            gemm(in, sa[k][j], ob, rows, Kp, 1, 0);
            in = ob; Kp = sa[k][j].N;
        }
        k_maxpool<<<(np * Kp + 255) / 256, 256, 0, stream>>>(in, ptsS[k+1], np, Kp);
    }

    // ---- FP stages ----
    auto run_fp = [&](const float* x1, const int* c1, int N1,
                      const float* x2, const int* c2, int N2,
                      const h16* p1, int C1, const h16* p2, int C2,
                      Lyr* L, int nl, h16* dst) {
        int Kp = C1 + C2; // all multiples of 32
        k_fp_build<<<(N1 + 3) / 4, dim3(32, 4), 0, stream>>>(x1, c1, N1, x2, c2, N2,
                                                             p1, C1, p2, C2, bufA, Kp);
        const h16* in = bufA;
        for (int j = 0; j < nl; ++j) {
            h16* ob = (j == nl - 1) ? dst : ((in == bufA) ? bufB : bufA);
            gemm(in, L[j], ob, N1, Kp, 1, 0);
            in = ob; Kp = L[j].N;
        }
    };
    run_fp(XYZ[3], clsL[3], NL[3], XYZ[4], clsL[4], NL[4], ptsS[3], 512, ptsS[4], 1024, fp[0], 2, ptsF3);
    run_fp(XYZ[2], clsL[2], NL[2], XYZ[3], clsL[3], NL[3], ptsS[2], 256, ptsF3,   512, fp[1], 2, ptsF2);
    run_fp(XYZ[1], clsL[1], NL[1], XYZ[2], clsL[2], NL[2], ptsS[1], 128, ptsF2,   256, fp[2], 2, ptsF1);
    run_fp(XYZ[0], clsL[0], NL[0], XYZ[1], clsL[1], NL[1], nullptr,   0, ptsF1,   128, fp[3], 3, l0feat);

    // ---- conv1 + population norm + relu ----
    gemm(l0feat, cv, Xbuf, 8192, 64, 0, 1);
    k_colstats<<<64, 256, 0, stream>>>(Xbuf, 8192, 64, stats);
    k_norm<<<(8192 * 64) / 256, 256, 0, stream>>>(Xbuf, stats, (float*)d_out, 8192, 64);
}
